// Block_48885317763452
// MI455X (gfx1250) — matrix operand order repaired and checked by reading, not compiled, not run on hardware
//
#include <hip/hip_runtime.h>
#include <hip/hip_bf16.h>
#include <cstdint>

#define B_   4096
#define J_   17
#define C_   512
#define K_   3
#define TOK_ 64
#define CH_  1024
#define H_   8
#define DH_  64
#define G_   8
#define M_   (B_*J_)          // 69632
#define EPSF 1e-5f

typedef __attribute__((ext_vector_type(16))) __bf16 v16bf;
typedef __attribute__((ext_vector_type(8)))  float  v8f;

#if defined(__has_builtin)
#  if __has_builtin(__builtin_amdgcn_global_load_async_to_lds_b128)
#    define ASYNC_LDS 1
#  else
#    define ASYNC_LDS 0
#  endif
#else
#  define ASYNC_LDS 0
#endif

// builtin expects: (int4 AS1*, int4 AS3*, int offset, int cpol)
typedef int v4i32 __attribute__((vector_size(16)));
typedef __attribute__((address_space(1))) v4i32* gv4i;
typedef __attribute__((address_space(3))) v4i32* lv4i;

template<int N>
static __device__ __forceinline__ void wait_async(){
#if defined(__has_builtin) && __has_builtin(__builtin_amdgcn_s_wait_asynccnt)
  __builtin_amdgcn_s_wait_asynccnt(N);
#else
  if (N == 0) asm volatile("s_wait_asynccnt 0x0" ::: "memory");
  else        asm volatile("s_wait_asynccnt 0x3" ::: "memory");
#endif
}

static __device__ __forceinline__ unsigned short f2bf(float f){
  uint32_t u = __builtin_bit_cast(uint32_t, f);
  uint32_t r = (u + 0x7FFFu + ((u >> 16) & 1u)) >> 16;
  return (unsigned short)r;
}
static __device__ __forceinline__ float bf2f(unsigned short h){
  return __builtin_bit_cast(float, (uint32_t)h << 16);
}
static __device__ __forceinline__ float gelu_exact(float x){
  return 0.5f * x * (1.0f + erff(x * 0.70710678118654752440f));
}

// ---------------------------------------------------------------- misc
__global__ __launch_bounds__(256) void k_f32_to_bf16(
    const float* __restrict__ s, unsigned short* __restrict__ d, int n){
  int i = blockIdx.x * 256 + threadIdx.x;
  if (i < n) d[i] = f2bf(s[i]);
}

// LayerNorm over joints (J axis), output bf16 in (B*J, C) layout
__global__ __launch_bounds__(256) void k_ln_joints(
    const float* __restrict__ x, const float* __restrict__ w,
    const float* __restrict__ bs, unsigned short* __restrict__ xln){
  int t = blockIdx.x * 256 + threadIdx.x;        // over B*C
  int c = t & (C_ - 1); int b = t >> 9;
  const float* xb = x + (size_t)b * J_ * C_ + c;
  float v[J_]; float s = 0.f;
  #pragma unroll
  for (int j = 0; j < J_; ++j){ v[j] = xb[(size_t)j * C_]; s += v[j]; }
  float mean = s * (1.0f / J_);
  float q = 0.f;
  #pragma unroll
  for (int j = 0; j < J_; ++j){ float d0 = v[j] - mean; q += d0 * d0; }
  float inv = rsqrtf(q * (1.0f / J_) + EPSF);
  #pragma unroll
  for (int j = 0; j < J_; ++j)
    xln[((size_t)b * J_ + j) * C_ + c] = f2bf((v[j] - mean) * inv * w[j] + bs[j]);
}

// ---------------------------------------------------------------- WMMA GEMM
// Y[M,N] = X[M,K] * W^T  (+bias, opt GELU).  X: M x K bf16 row-major,
// W: N x K bf16 row-major.  EPI: 0 = f32 out, 1 = bf16 out, 2 = bf16 + GELU.
// Tile: BM=128, BN=64, BK=32. 8 waves; each wave owns a 32x32 output patch
// (4 wmma per K-step). Double-buffered LDS; staging via
// GLOBAL_LOAD_ASYNC_TO_LDS_B128 (ASYNCcnt) when available, else b128
// register staging.
template<int EPI>
__global__ __launch_bounds__(256) void k_gemm_bf16(
    const unsigned short* __restrict__ A,
    const unsigned short* __restrict__ W,
    const float* __restrict__ bias,
    void* __restrict__ Out, int Msz, int Nsz, int Ksz){
  constexpr int BM = 128, BN = 64, BK = 32;
  __shared__ unsigned short As[2][BM * BK];   // 8 KB per buffer
  __shared__ unsigned short Bs[2][BN * BK];   // 4 KB per buffer
  const int tid  = threadIdx.x;
  const int wave = tid >> 5, lane = tid & 31;
  const int lr = lane & 15, hf = lane >> 4;
  const int mb = (wave & 3) << 5;           // wave's 32-row sub-block
  const int nb = (wave >> 2) << 5;          // wave's 32-col sub-block
  const int m0 = blockIdx.x * BM;
  const int n0 = blockIdx.y * BN;

  // ---- staging assignment (one b128 each; A gets two items per thread)
  const int ra0 = tid >> 2;                  // rows 0..63
  const int ra1 = ra0 + 64;                  // rows 64..127
  const int qa  = tid & 3;                   // 16-byte quad within 64-byte row
  const uint4* gA0 = reinterpret_cast<const uint4*>(A + (size_t)(m0 + ra0) * Ksz) + qa;
  const uint4* gA1 = reinterpret_cast<const uint4*>(A + (size_t)(m0 + ra1) * Ksz) + qa;
  const uint4* gB  = reinterpret_cast<const uint4*>(W + (size_t)(n0 + ra0) * Ksz) + qa;

  v8f acc00 = {}, acc01 = {}, acc10 = {}, acc11 = {};
  const int nsteps = Ksz / BK;

  // fragment helper data
  const int arow0 = mb + lr, arow1 = mb + 16 + lr;
  const int bcol0 = nb + lr, bcol1 = nb + 16 + lr;

#if ASYNC_LDS
  // per-thread LDS destinations (same offsets in both buffers)
  auto issue_tile = [&](int t, int buf){
    const uint4* sA0 = gA0 + (size_t)t * 4;   // +BK elements = 4 uint4
    const uint4* sA1 = gA1 + (size_t)t * 4;
    const uint4* sB  = gB  + (size_t)t * 4;
    __builtin_amdgcn_global_load_async_to_lds_b128(
        (gv4i)sA0, (lv4i)&As[buf][(ra0 * 4 + qa) * 8], 0, 0);
    __builtin_amdgcn_global_load_async_to_lds_b128(
        (gv4i)sA1, (lv4i)&As[buf][(ra1 * 4 + qa) * 8], 0, 0);
    __builtin_amdgcn_global_load_async_to_lds_b128(
        (gv4i)sB,  (lv4i)&Bs[buf][(ra0 * 4 + qa) * 8], 0, 0);
  };
  issue_tile(0, 0);
  if (nsteps > 1) issue_tile(1, 1);

  for (int s = 0; s < nsteps; ++s) {
    const int cur = s & 1;
    if (s + 1 < nsteps) wait_async<3>();   // tile s landed; s+1 may be in flight
    else                wait_async<0>();
    __syncthreads();

    union F { uint32_t u[8]; v16bf v; } a0, a1, b0, b1;
    const unsigned short* Ab = As[cur];
    const unsigned short* Bb = Bs[cur];
    #pragma unroll
    for (int v = 0; v < 8; ++v) {
      int kb = ((v & 3) << 1) + (hf << 3) + ((v >> 2) << 4);
      a0.u[v] = *reinterpret_cast<const uint32_t*>(&Ab[arow0 * BK + kb]);
      a1.u[v] = *reinterpret_cast<const uint32_t*>(&Ab[arow1 * BK + kb]);
    }
    #pragma unroll
    for (int v = 0; v < 8; ++v) {
      int kb = ((v & 3) << 1) + (hf << 3) + ((v >> 2) << 4);
      b0.u[v] = *reinterpret_cast<const uint32_t*>(&Bb[bcol0 * BK + kb]);
      b1.u[v] = *reinterpret_cast<const uint32_t*>(&Bb[bcol1 * BK + kb]);
    }
    acc00 = __builtin_amdgcn_wmma_f32_16x16x32_bf16(false, a0.v, false, b0.v,
                                                    (short)0, acc00, false, false);
    acc01 = __builtin_amdgcn_wmma_f32_16x16x32_bf16(false, a0.v, false, b1.v,
                                                    (short)0, acc01, false, false);
    acc10 = __builtin_amdgcn_wmma_f32_16x16x32_bf16(false, a1.v, false, b0.v,
                                                    (short)0, acc10, false, false);
    acc11 = __builtin_amdgcn_wmma_f32_16x16x32_bf16(false, a1.v, false, b1.v,
                                                    (short)0, acc11, false, false);
    __syncthreads();                        // everyone done reading buf cur
    if (s + 2 < nsteps) {
      issue_tile(s + 2, cur);               // refill the buffer just consumed
      __builtin_prefetch(gA0 + (size_t)(s + 3) * 4, 0, 1);
    }
  }
#else
  uint4 tA0 = gA0[0];
  uint4 tA1 = gA1[0];
  uint4 tB  = gB[0];
  for (int s = 0; s < nsteps; ++s) {
    const int cur = s & 1;
    reinterpret_cast<uint4*>(As[cur])[ra0 * 4 + qa] = tA0;
    reinterpret_cast<uint4*>(As[cur])[ra1 * 4 + qa] = tA1;
    reinterpret_cast<uint4*>(Bs[cur])[ra0 * 4 + qa] = tB;
    __syncthreads();

    if (s + 1 < nsteps) {
      gA0 += 4; gA1 += 4; gB += 4;
      tA0 = gA0[0]; tA1 = gA1[0]; tB = gB[0];
      if (s + 2 < nsteps) __builtin_prefetch(gA0 + 4, 0, 1);
    }

    union F { uint32_t u[8]; v16bf v; } a0, a1, b0, b1;
    const unsigned short* Ab = As[cur];
    const unsigned short* Bb = Bs[cur];
    #pragma unroll
    for (int v = 0; v < 8; ++v) {
      int kb = ((v & 3) << 1) + (hf << 3) + ((v >> 2) << 4);
      a0.u[v] = *reinterpret_cast<const uint32_t*>(&Ab[arow0 * BK + kb]);
      a1.u[v] = *reinterpret_cast<const uint32_t*>(&Ab[arow1 * BK + kb]);
    }
    #pragma unroll
    for (int v = 0; v < 8; ++v) {
      int kb = ((v & 3) << 1) + (hf << 3) + ((v >> 2) << 4);
      b0.u[v] = *reinterpret_cast<const uint32_t*>(&Bb[bcol0 * BK + kb]);
      b1.u[v] = *reinterpret_cast<const uint32_t*>(&Bb[bcol1 * BK + kb]);
    }
    acc00 = __builtin_amdgcn_wmma_f32_16x16x32_bf16(false, a0.v, false, b0.v,
                                                    (short)0, acc00, false, false);
    acc01 = __builtin_amdgcn_wmma_f32_16x16x32_bf16(false, a0.v, false, b1.v,
                                                    (short)0, acc01, false, false);
    acc10 = __builtin_amdgcn_wmma_f32_16x16x32_bf16(false, a1.v, false, b0.v,
                                                    (short)0, acc10, false, false);
    acc11 = __builtin_amdgcn_wmma_f32_16x16x32_bf16(false, a1.v, false, b1.v,
                                                    (short)0, acc11, false, false);
  }
#endif

  // ---- epilogue.  C/D layout: VGPR r -> M = r + 8*hf, N = lr
  #pragma unroll
  for (int t = 0; t < 4; ++t) {
    const v8f acc = (t == 0) ? acc00 : (t == 1) ? acc01 : (t == 2) ? acc10 : acc11;
    const int mt = t >> 1, nt = t & 1;
    const int mrow = m0 + mb + mt * 16 + hf * 8;
    const int ncol = n0 + nb + nt * 16 + lr;
    const float bv = bias[ncol];
    #pragma unroll
    for (int r = 0; r < 8; ++r) {
      float v0 = acc[r] + bv;
      if (EPI == 2) v0 = gelu_exact(v0);
      if (EPI == 0) ((float*)Out)[(size_t)(mrow + r) * Nsz + ncol] = v0;
      else ((unsigned short*)Out)[(size_t)(mrow + r) * Nsz + ncol] = f2bf(v0);
    }
  }
}

// ---------------------------------------------------------------- adjacency
// out[(b,w),c] = sum_{k,v} Y[(b,v), k*C+c] * adj[k,v,w]  (+ xj + x when FINAL)
template<int FINAL>
__global__ __launch_bounds__(256) void k_adj_contract(
    const unsigned short* __restrict__ Y, const float* __restrict__ adj,
    const float* __restrict__ xj, const float* __restrict__ x,
    float* __restrict__ out){
  int t = blockIdx.x * 256 + threadIdx.x;       // over M*C
  int c = t & (C_ - 1);
  int r = t >> 9;
  int w = r % J_;
  int b = r / J_;
  const unsigned short* Yb = Y + (size_t)b * J_ * (3 * C_) + c;
  float acc = 0.f;
  #pragma unroll
  for (int k = 0; k < K_; ++k) {
    #pragma unroll
    for (int v = 0; v < J_; ++v)
      acc += bf2f(Yb[(size_t)v * (3 * C_) + k * C_]) * adj[(k * J_ + v) * J_ + w];
  }
  if (FINAL) out[t] = acc + xj[t] + x[t];
  else       out[t] = acc;
}

// ---------------------------------------------------------------- dwconv + GN + GELU
__global__ __launch_bounds__(256) void k_dw_gn_gelu(
    const unsigned short* __restrict__ xln,
    const float* __restrict__ dww, const float* __restrict__ dwb,
    const float* __restrict__ gw,  const float* __restrict__ gb,
    unsigned short* __restrict__ hbf){
  const int GE = (C_ / G_) * J_;                // 1088
  __shared__ float sv[(C_ / G_) * J_];
  __shared__ float rs[256], rq[256];
  int bg = blockIdx.x; int g = bg & (G_ - 1); int b = bg >> 3;
  int c0 = g * (C_ / G_);
  int tid = threadIdx.x;
  float s = 0.f, q = 0.f;
  for (int idx = tid; idx < GE; idx += 256) {
    int cl = idx / J_, j = idx % J_;
    int c = c0 + cl;
    size_t base = (size_t)b * J_ * C_ + c;
    float xm = bf2f(xln[base + (size_t)j * C_]);
    float xa = (j > 0)      ? bf2f(xln[base + (size_t)(j - 1) * C_]) : 0.f;
    float xd = (j < J_ - 1) ? bf2f(xln[base + (size_t)(j + 1) * C_]) : 0.f;
    float hv = xa * dww[c * 3 + 0] + xm * dww[c * 3 + 1] + xd * dww[c * 3 + 2] + dwb[c];
    sv[idx] = hv; s += hv; q += hv * hv;
  }
  rs[tid] = s; rq[tid] = q;
  __syncthreads();
  for (int st = 128; st > 0; st >>= 1) {
    if (tid < st) { rs[tid] += rs[tid + st]; rq[tid] += rq[tid + st]; }
    __syncthreads();
  }
  float mean = rs[0] / GE;
  float var  = rq[0] / GE - mean * mean;
  float inv  = rsqrtf(var + EPSF);
  for (int idx = tid; idx < GE; idx += 256) {
    int cl = idx / J_, j = idx % J_;
    int c = c0 + cl;
    float val = (sv[idx] - mean) * inv * gw[c] + gb[c];
    hbf[((size_t)b * J_ + j) * C_ + c] = f2bf(gelu_exact(val));
  }
}

// ---------------------------------------------------------------- token MLP (J->TOK->J)
__global__ __launch_bounds__(256) void k_tokmlp(
    const float* __restrict__ pwo,
    const float* __restrict__ w1, const float* __restrict__ b1,
    const float* __restrict__ w2, const float* __restrict__ b2,
    float* __restrict__ tok){
  int t = blockIdx.x * 256 + threadIdx.x;       // over B*C
  int c = t & (C_ - 1); int b = t >> 9;
  const float* pb = pwo + (size_t)b * J_ * C_ + c;
  float hv[J_], ov[J_];
  #pragma unroll
  for (int j = 0; j < J_; ++j){ hv[j] = pb[(size_t)j * C_]; ov[j] = 0.f; }
  for (int tt = 0; tt < TOK_; ++tt) {
    float s = b1[tt];
    #pragma unroll
    for (int j = 0; j < J_; ++j) s += w1[tt * J_ + j] * hv[j];
    s = gelu_exact(s);
    #pragma unroll
    for (int j = 0; j < J_; ++j) ov[j] += w2[j * TOK_ + tt] * s;
  }
  #pragma unroll
  for (int j = 0; j < J_; ++j)
    tok[((size_t)b * J_ + j) * C_ + c] = ov[j] + b2[j];
}

// ---------------------------------------------------------------- attention per (b,h)
__global__ __launch_bounds__(128) void k_attn(
    const unsigned short* __restrict__ qkv,     // M x 1536 bf16
    unsigned short* __restrict__ sga){          // M x 512 bf16
  __shared__ float qs[DH_ * J_], ks[DH_ * J_], vs[DH_ * J_];
  __shared__ float S[J_ * J_];
  int bh = blockIdx.x; int h = bh & (H_ - 1); int b = bh >> 3;
  int tid = threadIdx.x;
  const unsigned short* base = qkv + (size_t)b * J_ * (3 * C_) + h * DH_;
  for (int idx = tid; idx < DH_ * J_; idx += 128) {
    int d = idx / J_, i = idx % J_;
    size_t o = (size_t)i * (3 * C_) + d;
    qs[idx] = bf2f(base[o]);
    ks[idx] = bf2f(base[o + C_]);
    vs[idx] = bf2f(base[o + 2 * C_]);
  }
  __syncthreads();
  for (int idx = tid; idx < J_ * J_; idx += 128) {
    int i = idx / J_, j = idx % J_;
    float a = 0.f;
    for (int d = 0; d < DH_; ++d) a += qs[d * J_ + i] * ks[d * J_ + j];
    S[idx] = a * 0.125f;                        // 1/sqrt(64)
  }
  __syncthreads();
  if (tid < J_) {
    int i = tid;
    float mx = -1e30f;
    for (int j = 0; j < J_; ++j) mx = fmaxf(mx, S[i * J_ + j]);
    float sum = 0.f;
    for (int j = 0; j < J_; ++j){ float e = expf(S[i * J_ + j] - mx); S[i * J_ + j] = e; sum += e; }
    float r = 1.f / sum;
    for (int j = 0; j < J_; ++j) S[i * J_ + j] *= r;
  }
  __syncthreads();
  for (int idx = tid; idx < DH_ * J_; idx += 128) {
    int d = idx / J_, i = idx % J_;
    float a = 0.f;
    for (int j = 0; j < J_; ++j) a += S[i * J_ + j] * vs[d * J_ + j];
    sga[((size_t)b * J_ + i) * C_ + h * DH_ + d] = f2bf(a);
  }
}

// ---------------------------------------------------------------- fuse + GN + GELU
__global__ __launch_bounds__(256) void k_fuse_gn_gelu(
    const float* __restrict__ p1, const float* __restrict__ p2,
    const float* __restrict__ p3,
    const float* __restrict__ gw, const float* __restrict__ gb,
    float* __restrict__ xj){
  const int GE = (C_ / G_) * J_;
  __shared__ float sv[(C_ / G_) * J_];
  __shared__ float rs[256], rq[256];
  int bg = blockIdx.x; int g = bg & (G_ - 1); int b = bg >> 3;
  int c0 = g * (C_ / G_);
  int tid = threadIdx.x;
  float s = 0.f, q = 0.f;
  for (int idx = tid; idx < GE; idx += 256) {
    int cl = idx / J_, j = idx % J_;
    size_t o = ((size_t)b * J_ + j) * C_ + c0 + cl;
    float v = p1[o] + p2[o] + p3[o];
    sv[idx] = v; s += v; q += v * v;
  }
  rs[tid] = s; rq[tid] = q;
  __syncthreads();
  for (int st = 128; st > 0; st >>= 1) {
    if (tid < st) { rs[tid] += rs[tid + st]; rq[tid] += rq[tid + st]; }
    __syncthreads();
  }
  float mean = rs[0] / GE;
  float var  = rq[0] / GE - mean * mean;
  float inv  = rsqrtf(var + EPSF);
  for (int idx = tid; idx < GE; idx += 256) {
    int cl = idx / J_, j = idx % J_;
    int c = c0 + cl;
    float val = (sv[idx] - mean) * inv * gw[c] + gb[c];
    xj[((size_t)b * J_ + j) * C_ + c] = gelu_exact(val);
  }
}

// ---------------------------------------------------------------- LN over channels of (xj + x)
__global__ __launch_bounds__(256) void k_ln_channels(
    const float* __restrict__ xj, const float* __restrict__ x,
    const float* __restrict__ w,  const float* __restrict__ bs,
    unsigned short* __restrict__ xn){
  __shared__ float rs[256], rq[256];
  int r = blockIdx.x;                            // row over M
  int tid = threadIdx.x;
  const float* a = xj + (size_t)r * C_;
  const float* b = x  + (size_t)r * C_;
  float t0 = a[tid] + b[tid];
  float t1 = a[tid + 256] + b[tid + 256];
  rs[tid] = t0 + t1; rq[tid] = t0 * t0 + t1 * t1;
  __syncthreads();
  for (int st = 128; st > 0; st >>= 1) {
    if (tid < st) { rs[tid] += rs[tid + st]; rq[tid] += rq[tid + st]; }
    __syncthreads();
  }
  float mean = rs[0] / C_;
  float var  = rq[0] / C_ - mean * mean;
  float inv  = rsqrtf(var + EPSF);
  xn[(size_t)r * C_ + tid]       = f2bf((t0 - mean) * inv * w[tid]       + bs[tid]);
  xn[(size_t)r * C_ + tid + 256] = f2bf((t1 - mean) * inv * w[tid + 256] + bs[tid + 256]);
}

// ---------------------------------------------------------------- launch
extern "C" void kernel_launch(void* const* d_in, const int* in_sizes, int n_in,
                              void* d_out, int out_size, void* d_ws, size_t ws_size,
                              hipStream_t stream) {
  const float* x     = (const float*)d_in[0];
  const float* adj   = (const float*)d_in[1];
  const float* n1w   = (const float*)d_in[2];
  const float* n1b   = (const float*)d_in[3];
  const float* gcn1w = (const float*)d_in[4];
  const float* gcn1b = (const float*)d_in[5];
  const float* dww   = (const float*)d_in[6];
  const float* dwb   = (const float*)d_in[7];
  const float* gn1w  = (const float*)d_in[8];
  const float* gn1b  = (const float*)d_in[9];
  const float* pww   = (const float*)d_in[10];
  const float* pwb   = (const float*)d_in[11];
  const float* m1w1  = (const float*)d_in[12];
  const float* m1b1  = (const float*)d_in[13];
  const float* m1w2  = (const float*)d_in[14];
  const float* m1b2  = (const float*)d_in[15];
  const float* qkvw  = (const float*)d_in[16];
  const float* qkvb  = (const float*)d_in[17];
  const float* projw = (const float*)d_in[18];
  const float* projb = (const float*)d_in[19];
  const float* gnfw  = (const float*)d_in[20];
  const float* gnfb  = (const float*)d_in[21];
  const float* n2w   = (const float*)d_in[22];
  const float* n2b   = (const float*)d_in[23];
  const float* m2w1  = (const float*)d_in[24];
  const float* m2b1  = (const float*)d_in[25];
  const float* m2w2  = (const float*)d_in[26];
  const float* m2b2  = (const float*)d_in[27];
  const float* gcn2w = (const float*)d_in[28];
  const float* gcn2b = (const float*)d_in[29];

  char* ws = (char*)d_ws;
  size_t off = 0;
  auto alloc = [&](size_t bytes){ size_t o = off; off += (bytes + 255) & ~(size_t)255; return o; };
  const size_t M = (size_t)M_;

  unsigned short* XLN = (unsigned short*)(ws + alloc(M * C_ * 2));       // later SGA
  unsigned short* WBF = (unsigned short*)(ws + alloc((size_t)3932160 * 2));
  unsigned short* YK  = (unsigned short*)(ws + alloc(M * 3 * C_ * 2));   // Y1 then Y2
  float*          A1  = (float*)(ws + alloc(M * C_ * 4));
  unsigned short* HBF = (unsigned short*)(ws + alloc(M * C_ * 2));       // later XN
  float*          PW  = (float*)(ws + alloc(M * C_ * 4));                // later PROJ
  float*          TOKB= (float*)(ws + alloc(M * C_ * 4));
  unsigned short* QKV = (unsigned short*)(ws + alloc(M * 3 * C_ * 2));   // later M1
  float*          XJ  = (float*)(ws + alloc(M * C_ * 4));
  unsigned short* M2B = (unsigned short*)(ws + alloc(M * C_ * 2));

  unsigned short* Wgcn1 = WBF;
  unsigned short* Wqkv  = WBF +  786432;
  unsigned short* Wgcn2 = WBF + 1572864;
  unsigned short* Wpw   = WBF + 2359296;
  unsigned short* Wproj = WBF + 2621440;
  unsigned short* Wm1   = WBF + 2883584;
  unsigned short* Wm2   = WBF + 3407872;

  auto cvt = [&](const float* s, unsigned short* d, int n){
    k_f32_to_bf16<<<(n + 255) / 256, 256, 0, stream>>>(s, d, n);
  };
  cvt(gcn1w, Wgcn1, 1536 * 512);
  cvt(qkvw,  Wqkv,  1536 * 512);
  cvt(gcn2w, Wgcn2, 1536 * 512);
  cvt(pww,   Wpw,    512 * 512);
  cvt(projw, Wproj,  512 * 512);
  cvt(m2w1,  Wm1,   1024 * 512);
  cvt(m2w2,  Wm2,    512 * 1024);

  // LN over joints -> XLN (bf16, BJC)
  k_ln_joints<<<(B_ * C_) / 256, 256, 0, stream>>>(x, n1w, n1b, XLN);

  dim3 gN1536(M_ / 128, 1536 / 64);
  dim3 gN512 (M_ / 128,  512 / 64);
  dim3 gN1024(M_ / 128, 1024 / 64);

  // branch 1: gcn1 GEMM + adjacency
  k_gemm_bf16<1><<<gN1536, 256, 0, stream>>>(XLN, Wgcn1, gcn1b, YK, M_, 1536, 512);
  k_adj_contract<0><<<(M_ * C_) / 256, 256, 0, stream>>>(YK, adj, nullptr, nullptr, A1);

  // branch 2: dwconv + GN + GELU, pw GEMM, token MLP
  k_dw_gn_gelu<<<B_ * G_, 256, 0, stream>>>(XLN, dww, dwb, gn1w, gn1b, HBF);
  k_gemm_bf16<0><<<gN512, 256, 0, stream>>>(HBF, Wpw, pwb, PW, M_, 512, 512);
  k_tokmlp<<<(B_ * C_) / 256, 256, 0, stream>>>(PW, m1w1, m1b1, m1w2, m1b2, TOKB);

  // branch 3: qkv GEMM, attention, proj GEMM
  k_gemm_bf16<1><<<gN1536, 256, 0, stream>>>(XLN, Wqkv, qkvb, QKV, M_, 1536, 512);
  unsigned short* SGA = XLN;                       // XLN no longer needed
  k_attn<<<B_ * H_, 128, 0, stream>>>(QKV, SGA);
  float* PROJ = PW;                                // PW consumed by tokmlp
  k_gemm_bf16<0><<<gN512, 256, 0, stream>>>(SGA, Wproj, projb, PROJ, M_, 512, 512);

  // fuse branches -> XJ; LN over channels -> XN
  k_fuse_gn_gelu<<<B_ * G_, 256, 0, stream>>>(PROJ, A1, TOKB, gnfw, gnfb, XJ);
  unsigned short* XN = HBF;                        // HBF consumed by pw GEMM
  k_ln_channels<<<M_, 256, 0, stream>>>(XJ, x, n2w, n2b, XN);

  // mlp2: fc1 (GELU) -> M1, fc2 -> M2B
  unsigned short* M1 = QKV;                        // QKV consumed by attention
  k_gemm_bf16<2><<<gN1024, 256, 0, stream>>>(XN, Wm1, m2b1, M1, M_, 1024, 512);
  k_gemm_bf16<1><<<gN512, 256, 0, stream>>>(M1, Wm2, m2b2, M2B, M_, 512, 1024);

  // gcn2 GEMM + final adjacency with residuals -> out
  k_gemm_bf16<1><<<gN1536, 256, 0, stream>>>(M2B, Wgcn2, gcn2b, YK, M_, 1536, 512);
  k_adj_contract<1><<<(M_ * C_) / 256, 256, 0, stream>>>(YK, adj, XJ, x, (float*)d_out);
}
